// EncoderBlock_81595788689914
// MI455X (gfx1250) — matrix operand order repaired and checked by reading, not compiled, not run on hardware
//
#include <hip/hip_runtime.h>
#include <hip/hip_bf16.h>
#include <math.h>

typedef __attribute__((ext_vector_type(16))) __bf16 v16bf;
typedef __attribute__((ext_vector_type(8)))  __bf16 v8bf;
typedef __attribute__((ext_vector_type(8)))  float  v8f;
typedef __attribute__((ext_vector_type(4)))  int    v4i;

union bf16x16u { v16bf v; v8bf h[2]; };

static __device__ __forceinline__ v8f wmma_bf16(v16bf a, v16bf b, v8f c) {
    // D = A(16x32 bf16) x B(32x16 bf16) + C(16x16 f32)
    return __builtin_amdgcn_wmma_f32_16x16x32_bf16(
        false, a, false, b, (short)0, c, false, false);
}

#if __has_builtin(__builtin_amdgcn_global_load_async_to_lds_b128) && \
    __has_builtin(__builtin_amdgcn_s_wait_asynccnt)
#define HAVE_ASYNC_LDS 1
#else
#define HAVE_ASYNC_LDS 0
#endif

#if HAVE_ASYNC_LDS
static __device__ __forceinline__ void async_copy16(const __bf16* g, __bf16* l) {
    __builtin_amdgcn_global_load_async_to_lds_b128(
        (__attribute__((address_space(1))) v4i*)(g),
        (__attribute__((address_space(3))) v4i*)(l),
        /*offset=*/0, /*cpol=*/0);
}
#endif

// ---------------------------------------------------------------------------
// Weight convert + transpose: W (K x N, f32, row-major) -> WT (N x K, bf16)
// ---------------------------------------------------------------------------
__global__ __launch_bounds__(256)
void enc_cvt_transpose_w(const float* __restrict__ W, __bf16* __restrict__ WT,
                         int K, int N) {
    size_t i = (size_t)blockIdx.x * 256 + threadIdx.x;
    size_t total = (size_t)K * N;
    if (i >= total) return;
    int k = (int)(i / N);
    int n = (int)(i % N);
    WT[(size_t)n * K + k] = (__bf16)W[i];
}

// ---------------------------------------------------------------------------
// LayerNorm (reference semantics: var over n-1, alpha*(x-mean)/(std+eps)+beta)
// ---------------------------------------------------------------------------
__global__ __launch_bounds__(256)
void enc_layernorm(const float* __restrict__ x, __bf16* __restrict__ out,
                   const float* __restrict__ alpha_p, const float* __restrict__ beta_p) {
    __shared__ float red[8];
    const int row = blockIdx.x;
    const int tid = threadIdx.x;
    const float* xr = x + (size_t)row * 1024;

    float lv[4];
    float s = 0.f;
#pragma unroll
    for (int i = 0; i < 4; ++i) { lv[i] = xr[tid + 256 * i]; s += lv[i]; }
#pragma unroll
    for (int m = 16; m >= 1; m >>= 1) s += __shfl_xor(s, m, 32);
    if ((tid & 31) == 0) red[tid >> 5] = s;
    __syncthreads();
    float tot = 0.f;
#pragma unroll
    for (int i = 0; i < 8; ++i) tot += red[i];
    const float mean = tot * (1.0f / 1024.0f);
    __syncthreads();

    float s2 = 0.f;
#pragma unroll
    for (int i = 0; i < 4; ++i) { float d = lv[i] - mean; s2 += d * d; }
#pragma unroll
    for (int m = 16; m >= 1; m >>= 1) s2 += __shfl_xor(s2, m, 32);
    if ((tid & 31) == 0) red[tid >> 5] = s2;
    __syncthreads();
    float tv = 0.f;
#pragma unroll
    for (int i = 0; i < 8; ++i) tv += red[i];

    const float var = tv * (1.0f / 1023.0f);
    const float inv = 1.0f / (sqrtf(var) + 1e-6f);
    const float a = alpha_p[0], b = beta_p[0];
#pragma unroll
    for (int i = 0; i < 4; ++i)
        out[(size_t)row * 1024 + tid + 256 * i] = (__bf16)(a * (lv[i] - mean) * inv + b);
}

// ---------------------------------------------------------------------------
// bf16 WMMA GEMM: C(MxN,f32) = A(MxK,bf16) @ B + bias, B given as BT (NxK).
// Block = 256 threads (8 waves); macro-tile 128(M) x 128(N); K step 32.
// Wave (wm 0..3, wn 0..1) owns 32x64 = 2x4 WMMA tiles (8 WMMAs / K-step).
// Staging: double-buffered GLOBAL_LOAD_ASYNC_TO_LDS_B128 (ASYNCcnt) when
// available; otherwise register-prefetch software pipeline.
// ---------------------------------------------------------------------------
#define GT_M 128
#define GT_N 128
#define GT_K 32
#define LDS_STRIDE 40   // 32 + 8 pad -> 80B rows, 16B aligned

__global__ __launch_bounds__(256)
void enc_gemm_bf16(const __bf16* __restrict__ A, const __bf16* __restrict__ BT,
                   const float* __restrict__ bias, const float* __restrict__ resid,
                   void* __restrict__ outp, int M, int N, int K,
                   int do_relu, int out_is_bf16) {
    __shared__ __bf16 sA[2][GT_M][LDS_STRIDE];
    __shared__ __bf16 sB[2][GT_N][LDS_STRIDE];

    const int tid  = threadIdx.x;
    const int wave = tid >> 5;
    const int lane = tid & 31;
    const int c    = lane & 15;
    const int half = lane >> 4;
    const int wm   = wave & 3;
    const int wn   = wave >> 2;
    const int m0   = blockIdx.x * GT_M;
    const int n0   = blockIdx.y * GT_N;

    // per-thread staging coordinates (two 16B chunks each for A and B)
    int srow[2], scol[2];
#pragma unroll
    for (int r = 0; r < 2; ++r) {
        const int linear = (r * 256 + tid) * 8;   // 4096 bf16 per tile
        srow[r] = linear >> 5;                    // 0..127
        scol[r] = linear & 31;
    }

    v8f acc[2][4] = {};
    const int tiles = K / GT_K;

    auto compute = [&](int cur) {
        bf16x16u bfr[4];
#pragma unroll
        for (int tj = 0; tj < 4; ++tj) {
            const int bn = wn * 64 + tj * 16 + c;
            bfr[tj].h[0] = *(const v8bf*)(&sB[cur][bn][half * 8]);
            bfr[tj].h[1] = *(const v8bf*)(&sB[cur][bn][16 + half * 8]);
        }
#pragma unroll
        for (int ti = 0; ti < 2; ++ti) {
            bf16x16u af;
            const int ar = wm * 32 + ti * 16 + c;
            af.h[0] = *(const v8bf*)(&sA[cur][ar][half * 8]);
            af.h[1] = *(const v8bf*)(&sA[cur][ar][16 + half * 8]);
#pragma unroll
            for (int tj = 0; tj < 4; ++tj)
                acc[ti][tj] = wmma_bf16(af.v, bfr[tj].v, acc[ti][tj]);
        }
    };

#if HAVE_ASYNC_LDS
    auto stage_async = [&](int buf, int k0) {
#pragma unroll
        for (int r = 0; r < 2; ++r) {
            async_copy16(A  + (size_t)(m0 + srow[r]) * K + k0 + scol[r],
                         &sA[buf][srow[r]][scol[r]]);
            async_copy16(BT + (size_t)(n0 + srow[r]) * K + k0 + scol[r],
                         &sB[buf][srow[r]][scol[r]]);
        }
    };
    stage_async(0, 0);
    for (int t = 0; t < tiles; ++t) {
        const int cur = t & 1;
        if (t + 1 < tiles) {
            stage_async(cur ^ 1, (t + 1) * GT_K);
            __builtin_amdgcn_s_wait_asynccnt(4);   // tile t's 4 copies done
        } else {
            __builtin_amdgcn_s_wait_asynccnt(0);
        }
        __syncthreads();
        compute(cur);
        __syncthreads();
    }
#else
    v8bf pa[2], pb[2];
    auto fetch = [&](int k0) {
#pragma unroll
        for (int r = 0; r < 2; ++r) {
            pa[r] = *(const v8bf*)(A  + (size_t)(m0 + srow[r]) * K + k0 + scol[r]);
            pb[r] = *(const v8bf*)(BT + (size_t)(n0 + srow[r]) * K + k0 + scol[r]);
        }
    };
    fetch(0);
    for (int t = 0; t < tiles; ++t) {
#pragma unroll
        for (int r = 0; r < 2; ++r) {
            *(v8bf*)(&sA[0][srow[r]][scol[r]]) = pa[r];
            *(v8bf*)(&sB[0][srow[r]][scol[r]]) = pb[r];
        }
        __syncthreads();
        if (t + 1 < tiles) fetch((t + 1) * GT_K);   // overlap with compute
        compute(0);
        __syncthreads();
    }
#endif

    // Epilogue: C element (v, lane) -> row v + 8*half, col c
#pragma unroll
    for (int ti = 0; ti < 2; ++ti) {
#pragma unroll
        for (int tj = 0; tj < 4; ++tj) {
            const int col = n0 + wn * 64 + tj * 16 + c;
            const float bv = bias ? bias[col] : 0.0f;
#pragma unroll
            for (int v = 0; v < 8; ++v) {
                const int row = m0 + wm * 32 + ti * 16 + v + 8 * half;
                float val = acc[ti][tj][v] + bv;
                if (do_relu) val = fmaxf(val, 0.0f);
                const size_t idx = (size_t)row * N + col;
                if (resid) val += resid[idx];
                if (out_is_bf16) ((__bf16*)outp)[idx] = (__bf16)val;
                else             ((float*)outp)[idx] = val;
            }
        }
    }
}

// ---------------------------------------------------------------------------
// V transpose: v [B*S][1024] bf16 -> vt [B][H][64][S] bf16 (per-head d-major)
// ---------------------------------------------------------------------------
__global__ __launch_bounds__(256)
void enc_transpose_v(const __bf16* __restrict__ v, __bf16* __restrict__ vt) {
    size_t i = (size_t)blockIdx.x * 256 + threadIdx.x;
    const size_t total = (size_t)2 * 2048 * 1024;
    if (i >= total) return;
    const int t = (int)(i >> 10);      // token 0..4095
    const int cd = (int)(i & 1023);    // h*64 + d
    const int b = t >> 11;
    const int s = t & 2047;
    vt[((size_t)b * 1024 + cd) * 2048 + s] = v[i];
}

// ---------------------------------------------------------------------------
// Flash attention: 1 wave per (qtile16, head, batch). 32-key inner tiles.
// ---------------------------------------------------------------------------
__global__ __launch_bounds__(32)
void enc_flash_attn(const __bf16* __restrict__ q, const __bf16* __restrict__ k,
                    const __bf16* __restrict__ vt, const int* __restrict__ mask,
                    __bf16* __restrict__ ctx) {
    const int S = 2048, H = 16, DK = 64, DM = 1024;
    const int lane = threadIdx.x;
    const int c = lane & 15;
    const int half = lane >> 4;
    const int qt = blockIdx.x;
    const int h  = blockIdx.y;
    const int b  = blockIdx.z;
    const int m0 = qt * 16;

    __shared__ __bf16 pLds[16][LDS_STRIDE];

    bf16x16u qf[2];
    {
        const __bf16* qrow = q + (size_t)(b * S + m0 + c) * DM + h * DK;
#pragma unroll
        for (int dh = 0; dh < 2; ++dh) {
            qf[dh].h[0] = *(const v8bf*)(qrow + dh * 32 + half * 8);
            qf[dh].h[1] = *(const v8bf*)(qrow + dh * 32 + 16 + half * 8);
        }
    }

    v8f o[4] = {};
    float mrow[8], lrow[8];
#pragma unroll
    for (int v = 0; v < 8; ++v) { mrow[v] = -3.0e38f; lrow[v] = 0.0f; }

    const float scale = 0.125f;               // 1/sqrt(64)
    const float L2E = 1.44269504088896340736f;

    for (int kv0 = 0; kv0 < S; kv0 += 32) {
        v8f sacc[2] = {};
#pragma unroll
        for (int t = 0; t < 2; ++t) {
            const __bf16* krow = k + (size_t)(b * S + kv0 + t * 16 + c) * DM + h * DK;
#pragma unroll
            for (int dh = 0; dh < 2; ++dh) {
                bf16x16u kf;
                kf.h[0] = *(const v8bf*)(krow + dh * 32 + half * 8);
                kf.h[1] = *(const v8bf*)(krow + dh * 32 + 16 + half * 8);
                sacc[t] = wmma_bf16(qf[dh].v, kf.v, sacc[t]);
            }
        }

        const int mk0 = mask[b * S + kv0 + c];
        const int mk1 = mask[b * S + kv0 + 16 + c];
        float mnew[8];
#pragma unroll
        for (int v = 0; v < 8; ++v) {
            float s0 = sacc[0][v] * scale; if (mk0 == 0) s0 = -1.0e9f;
            float s1 = sacc[1][v] * scale; if (mk1 == 0) s1 = -1.0e9f;
            sacc[0][v] = s0; sacc[1][v] = s1;
            mnew[v] = fmaxf(s0, s1);
        }
#pragma unroll
        for (int m = 8; m >= 1; m >>= 1)
#pragma unroll
            for (int v = 0; v < 8; ++v)
                mnew[v] = fmaxf(mnew[v], __shfl_xor(mnew[v], m, 32));

        float corr[8], rsum[8];
#pragma unroll
        for (int v = 0; v < 8; ++v) {
            mnew[v] = fmaxf(mnew[v], mrow[v]);
            corr[v] = exp2f((mrow[v] - mnew[v]) * L2E);
            mrow[v] = mnew[v];
            float p0 = exp2f((sacc[0][v] - mnew[v]) * L2E);
            float p1 = exp2f((sacc[1][v] - mnew[v]) * L2E);
            sacc[0][v] = p0; sacc[1][v] = p1;
            rsum[v] = p0 + p1;
        }
#pragma unroll
        for (int m = 8; m >= 1; m >>= 1)
#pragma unroll
            for (int v = 0; v < 8; ++v)
                rsum[v] += __shfl_xor(rsum[v], m, 32);
#pragma unroll
        for (int v = 0; v < 8; ++v) lrow[v] = lrow[v] * corr[v] + rsum[v];
#pragma unroll
        for (int t = 0; t < 4; ++t)
#pragma unroll
            for (int v = 0; v < 8; ++v) o[t][v] *= corr[v];

        // redistribute P: C-layout -> A-layout via LDS
        __syncthreads();
#pragma unroll
        for (int t = 0; t < 2; ++t)
#pragma unroll
            for (int v = 0; v < 8; ++v)
                pLds[v + 8 * half][t * 16 + c] = (__bf16)sacc[t][v];
        __syncthreads();
        bf16x16u pf;
        pf.h[0] = *(const v8bf*)(&pLds[c][half * 8]);
        pf.h[1] = *(const v8bf*)(&pLds[c][16 + half * 8]);

        const __bf16* vbase = vt + ((size_t)(b * H + h) * DK) * S + kv0 + half * 8;
#pragma unroll
        for (int tj = 0; tj < 4; ++tj) {
            bf16x16u vf;
            const __bf16* vrow = vbase + (size_t)(tj * 16 + c) * S;
            vf.h[0] = *(const v8bf*)(vrow);
            vf.h[1] = *(const v8bf*)(vrow + 16);
            o[tj] = wmma_bf16(pf.v, vf.v, o[tj]);
        }
    }

#pragma unroll
    for (int tj = 0; tj < 4; ++tj) {
#pragma unroll
        for (int v = 0; v < 8; ++v) {
            const float val = o[tj][v] / lrow[v];
            const int row = m0 + v + 8 * half;
            const int col = h * DK + tj * 16 + c;
            ctx[(size_t)(b * S + row) * DM + col] = (__bf16)val;
        }
    }
}

// ---------------------------------------------------------------------------
// Host-side orchestration
// ---------------------------------------------------------------------------
extern "C" void kernel_launch(void* const* d_in, const int* in_sizes, int n_in,
                              void* d_out, int out_size, void* d_ws, size_t ws_size,
                              hipStream_t stream) {
    (void)in_sizes; (void)n_in; (void)out_size; (void)ws_size;
    const float* x     = (const float*)d_in[0];
    const int*   mask  = (const int*)d_in[1];
    const float* wq    = (const float*)d_in[2];
    const float* bq    = (const float*)d_in[3];
    const float* wk    = (const float*)d_in[4];
    const float* bk    = (const float*)d_in[5];
    const float* wv    = (const float*)d_in[6];
    const float* bv    = (const float*)d_in[7];
    const float* wo    = (const float*)d_in[8];
    const float* bo    = (const float*)d_in[9];
    const float* w1    = (const float*)d_in[10];
    const float* b1    = (const float*)d_in[11];
    const float* w2    = (const float*)d_in[12];
    const float* b2    = (const float*)d_in[13];
    const float* ln1_a = (const float*)d_in[14];
    const float* ln1_b = (const float*)d_in[15];
    const float* ln2_a = (const float*)d_in[16];
    const float* ln2_b = (const float*)d_in[17];
    float* out = (float*)d_out;

    const int T = 4096;          // B*S tokens
    const int DM = 1024, DFF = 4096;

    char* ws = (char*)d_ws;
    size_t off = 0;
    auto carve = [&](size_t bytes) -> void* {
        void* p = ws + off;
        off += (bytes + 255) & ~(size_t)255;
        return p;
    };
    __bf16* wqT = (__bf16*)carve((size_t)DM * DM * 2);
    __bf16* wkT = (__bf16*)carve((size_t)DM * DM * 2);
    __bf16* wvT = (__bf16*)carve((size_t)DM * DM * 2);
    __bf16* woT = (__bf16*)carve((size_t)DM * DM * 2);
    __bf16* w1T = (__bf16*)carve((size_t)DM * DFF * 2);
    __bf16* w2T = (__bf16*)carve((size_t)DFF * DM * 2);
    __bf16* xn  = (__bf16*)carve((size_t)T * DM * 2);
    __bf16* qb  = (__bf16*)carve((size_t)T * DM * 2);
    __bf16* kb  = (__bf16*)carve((size_t)T * DM * 2);
    __bf16* vb  = (__bf16*)carve((size_t)T * DM * 2);
    __bf16* vtb = (__bf16*)carve((size_t)T * DM * 2);
    __bf16* ctx = (__bf16*)carve((size_t)T * DM * 2);
    float*  x2  = (float*)carve((size_t)T * DM * 4);
    __bf16* xn2 = (__bf16*)carve((size_t)T * DM * 2);
    __bf16* ff1 = (__bf16*)carve((size_t)T * DFF * 2);

    // Weights -> bf16, transposed
    enc_cvt_transpose_w<<<(DM * DM) / 256, 256, 0, stream>>>(wq, wqT, DM, DM);
    enc_cvt_transpose_w<<<(DM * DM) / 256, 256, 0, stream>>>(wk, wkT, DM, DM);
    enc_cvt_transpose_w<<<(DM * DM) / 256, 256, 0, stream>>>(wv, wvT, DM, DM);
    enc_cvt_transpose_w<<<(DM * DM) / 256, 256, 0, stream>>>(wo, woT, DM, DM);
    enc_cvt_transpose_w<<<(DM * DFF) / 256, 256, 0, stream>>>(w1, w1T, DM, DFF);
    enc_cvt_transpose_w<<<(DFF * DM) / 256, 256, 0, stream>>>(w2, w2T, DFF, DM);

    // LN1
    enc_layernorm<<<T, 256, 0, stream>>>(x, xn, ln1_a, ln1_b);

    // QKV projections
    dim3 g1(T / GT_M, DM / GT_N);
    enc_gemm_bf16<<<g1, 256, 0, stream>>>(xn, wqT, bq, nullptr, qb, T, DM, DM, 0, 1);
    enc_gemm_bf16<<<g1, 256, 0, stream>>>(xn, wkT, bk, nullptr, kb, T, DM, DM, 0, 1);
    enc_gemm_bf16<<<g1, 256, 0, stream>>>(xn, wvT, bv, nullptr, vb, T, DM, DM, 0, 1);

    // V -> V^T (per head, d-major)
    enc_transpose_v<<<(T * DM) / 256, 256, 0, stream>>>(vb, vtb);

    // Attention
    dim3 ga(2048 / 16, 16, 2);
    enc_flash_attn<<<ga, 32, 0, stream>>>(qb, kb, vtb, mask, ctx);

    // Output projection + residual -> x2 (f32)
    enc_gemm_bf16<<<g1, 256, 0, stream>>>(ctx, woT, bo, x, x2, T, DM, DM, 0, 0);

    // LN2
    enc_layernorm<<<T, 256, 0, stream>>>(x2, xn2, ln2_a, ln2_b);

    // FFN
    dim3 g2(T / GT_M, DFF / GT_N);
    enc_gemm_bf16<<<g2, 256, 0, stream>>>(xn2, w1T, b1, nullptr, ff1, T, DFF, DM, 1, 1);
    enc_gemm_bf16<<<g1, 256, 0, stream>>>(ff1, w2T, b2, x2, out, T, DM, DFF, 0, 0);
}
